// ProposalLayer_27711128994381
// MI455X (gfx1250) — compile-verified, matrix operand order checked by reading, not run on hardware
//
#include <hip/hip_runtime.h>
#include <hip/hip_bf16.h>
#include <math.h>

typedef __attribute__((ext_vector_type(16))) _Float16 v16h;
typedef __attribute__((ext_vector_type(8)))  float    v8f;

#define BB   16
#define NN   16384
#define CC   76
#define PRE0 358
#define PRE1 154
#define POST0 89
#define POST1 39
#define NMS_PRE 512
#define NMS_POST 128
#define NMS_THR 0.85f
#define TWO_PI 6.283185307179586f
#define APC   0.5235987755982988f   /* 2*pi/12 */

/* ---------------- Kernel 1: decode boxes (memory-bound streaming pass) -------- */
__global__ void decode_kernel(const float* __restrict__ reg,
                              const float* __restrict__ xyz,
                              const float* __restrict__ mean_size,
                              float* __restrict__ boxes) {
  int idx = blockIdx.x * blockDim.x + threadIdx.x;
  if (idx >= BB * NN) return;
  const float* rg = reg + (size_t)idx * CC;
  __builtin_prefetch(rg + CC, 0, 1);   // global_prefetch_b8: pull next row toward L2

  // argmax over 12-wide bins (first max wins, matching jnp.argmax)
  int xb = 0, zb = 0, rb = 0;
  float xv = rg[0], zv = rg[12], rv = rg[49];
  #pragma unroll
  for (int i = 1; i < 12; ++i) {
    float a = rg[i];      if (a > xv) { xv = a; xb = i; }
    float b = rg[12 + i]; if (b > zv) { zv = b; zb = i; }
    float c = rg[49 + i]; if (c > rv) { rv = c; rb = i; }
  }
  float x_res = rg[24 + xb] * 0.5f;
  float z_res = rg[36 + zb] * 0.5f;
  float px = (float)xb * 0.5f + 0.25f - 3.0f + x_res + xyz[idx * 3 + 0];
  float pz = (float)zb * 0.5f + 0.25f - 3.0f + z_res + xyz[idx * 3 + 2];

  float ry_res = rg[61 + rb];
  float ry = fmodf((float)rb * APC + ry_res * (APC * 0.5f), TWO_PI);
  if (ry < 0.0f) ry += TWO_PI;           // python % is non-negative
  if (ry > 3.14159265358979f) ry -= TWO_PI;

  float m0 = mean_size[0], m1 = mean_size[1], m2 = mean_size[2];
  float h = rg[73] * m0 + m0;
  float w = rg[74] * m1 + m1;
  float l = rg[75] * m2 + m2;
  float py = xyz[idx * 3 + 1] + rg[48] + 0.5f * h;   // includes props[...,1] += h/2

  float* o = boxes + (size_t)idx * 7;
  o[0] = px; o[1] = py; o[2] = pz; o[3] = h; o[4] = w; o[5] = l; o[6] = ry;
}

/* ---------------- Kernel 2: per (batch,range) top-k via LDS bitonic sort ------ */
/* 16384 x u64 keys = 128 KB dynamic LDS: fits in CDNA5's 320KB WGP LDS.        */
__global__ void topk_kernel(const float* __restrict__ scores,
                            const float* __restrict__ boxes,
                            float* __restrict__ sel_boxes,
                            float* __restrict__ sel_scores,
                            unsigned* __restrict__ sel_valid) {
  extern __shared__ unsigned long long s_keys[];
  const int b = blockIdx.x >> 1;
  const int r = blockIdx.x & 1;
  const float lo = r ? 40.0f : 0.0f;
  const float hi = r ? 80.0f : 40.0f;
  const int pre = r ? PRE1 : PRE0;
  const int tid = threadIdx.x;

  for (int i = tid; i < NN; i += 1024) {
    float s = scores[b * NN + i];
    float z = boxes[((size_t)(b * NN + i)) * 7 + 2];
    bool in = (z > lo) && (z <= hi);
    unsigned u = __float_as_uint(s);
    unsigned fk = (u & 0x80000000u) ? ~u : (u | 0x80000000u);   // order-preserving key
    s_keys[i] = in ? ((((unsigned long long)fk) << 32) | (unsigned)(~(unsigned)i)) : 0ull;
  }
  __syncthreads();

  // bitonic sort, descending
  for (unsigned k = 2; k <= (unsigned)NN; k <<= 1) {
    for (unsigned j = k >> 1; j > 0; j >>= 1) {
      for (unsigned i = tid; i < (unsigned)NN; i += 1024) {
        unsigned ixj = i ^ j;
        if (ixj > i) {
          bool up = ((i & k) == 0);
          unsigned long long A = s_keys[i], B = s_keys[ixj];
          bool sw = up ? (A < B) : (A > B);
          if (sw) { s_keys[i] = B; s_keys[ixj] = A; }
        }
      }
      __syncthreads();
    }
  }

  for (int i = tid; i < NMS_PRE; i += 1024) {
    unsigned long long k = s_keys[i];
    bool valid = (i < pre) && (k != 0ull);
    int src = valid ? (int)(~(unsigned)k) : 0;
    int dst = (b * 2 + r) * NMS_PRE + i;
    const float* bx = boxes + ((size_t)(b * NN + src)) * 7;
    #pragma unroll
    for (int c = 0; c < 7; ++c) sel_boxes[(size_t)dst * 7 + c] = bx[c];
    sel_scores[dst] = scores[b * NN + src];
    sel_valid[dst]  = valid ? 1u : 0u;
  }
}

/* ---- rotated-box IoU (Sutherland-Hodgman clip); cos/sin precomputed per box -- */
__device__ __forceinline__ void corners_cs(float cx, float cy, float w, float h,
                                           float c, float s, float* px, float* py) {
  const float lx[4] = {0.5f, -0.5f, -0.5f, 0.5f};
  const float ly[4] = {0.5f, 0.5f, -0.5f, -0.5f};
  #pragma unroll
  for (int i = 0; i < 4; ++i) {
    float ax = lx[i] * w, ay = ly[i] * h;
    px[i] = cx + c * ax - s * ay;
    py[i] = cy + s * ax + c * ay;
  }
}

__device__ float pair_iou_cs(float cx1, float cy1, float w1, float h1, float c1, float s1,
                             float cx2, float cy2, float w2, float h2, float c2, float s2) {
  float c1x[4], c1y[4], c2x[4], c2y[4];
  corners_cs(cx1, cy1, w1, h1, c1, s1, c1x, c1y);
  corners_cs(cx2, cy2, w2, h2, c2, s2, c2x, c2y);
  float px[9], py[9];
  #pragma unroll
  for (int i = 0; i < 9; ++i) { px[i] = 0.0f; py[i] = 0.0f; }
  #pragma unroll
  for (int i = 0; i < 4; ++i) { px[i] = c1x[i]; py[i] = c1y[i]; }
  int n = 4;
  #pragma unroll
  for (int e = 0; e < 4; ++e) {
    float ax = c2x[e], ay = c2y[e];
    float bx = c2x[(e + 1) & 3], by = c2y[(e + 1) & 3];
    float d[8];
    #pragma unroll
    for (int i = 0; i < 8; ++i)
      d[i] = (bx - ax) * (py[i] - ay) - (by - ay) * (px[i] - ax);
    float ox[9], oy[9];
    #pragma unroll
    for (int i = 0; i < 9; ++i) { ox[i] = 0.0f; oy[i] = 0.0f; }
    int cnt = 0;
    int nn = n > 1 ? n : 1;
    #pragma unroll
    for (int i = 0; i < 8; ++i) {
      int j = (i + 1) % nn;
      bool act = i < n;
      float dc = d[i], dn = d[j];
      bool keep = act && (dc >= 0.0f);
      if (keep) { int t = cnt < 8 ? cnt : 8; ox[t] = px[i]; oy[t] = py[i]; cnt++; }
      bool crossed = act && ((dc >= 0.0f) != (dn >= 0.0f));
      if (crossed) {
        float den = dc - dn;
        float t = dc / (fabsf(den) > 1e-12f ? den : 1.0f);
        int q = cnt < 8 ? cnt : 8;
        ox[q] = px[i] + t * (px[j] - px[i]);
        oy[q] = py[i] + t * (py[j] - py[i]);
        cnt++;
      }
    }
    #pragma unroll
    for (int i = 0; i < 9; ++i) { px[i] = ox[i]; py[i] = oy[i]; }
    n = cnt;
  }
  float inter = 0.0f;
  if (n >= 3) {
    int nn = n > 1 ? n : 1;
    float s = 0.0f;
    #pragma unroll
    for (int i = 0; i < 8; ++i) {
      if (i < n) {
        int nx = (i + 1) % nn;
        s += px[i] * py[nx] - px[nx] * py[i];
      }
    }
    inter = 0.5f * fabsf(s);
  }
  float a1 = w1 * h1, a2 = w2 * h2;
  return inter / fmaxf(a1 + a2 - inter, 1e-8f);
}

/* ---------------- Kernel 3: WMMA distance prefilter + IoU + greedy NMS -------- */
__global__ void nms_kernel(const float* __restrict__ sel_boxes,
                           const float* __restrict__ sel_scores,
                           const unsigned* __restrict__ sel_valid,
                           float* __restrict__ stage_boxes,
                           float* __restrict__ stage_scores,
                           int* __restrict__ counts) {
  const int b = blockIdx.x >> 1;
  const int r = blockIdx.x & 1;
  const int pre  = r ? PRE1 : PRE0;
  const int post = r ? POST1 : POST0;
  const int part = b * 2 + r;
  const int tid = threadIdx.x;

  __shared__ float s_cx[NMS_PRE], s_cz[NMS_PRE];   // BEV center
  __shared__ float s_w[NMS_PRE],  s_h[NMS_PRE];    // BEV dims (h_dim, l_dim)
  __shared__ float s_co[NMS_PRE], s_si[NMS_PRE];   // cos/sin(ry), computed once/box
  __shared__ float s_sq[NMS_PRE];                  // |c|^2
  __shared__ float s_rad[NMS_PRE];                 // circumscribed radius
  __shared__ unsigned s_val[NMS_PRE];
  __shared__ unsigned s_adj[NMS_PRE][16];          // iou>thr adjacency bitmask

  for (int i = tid; i < NMS_PRE; i += blockDim.x) {
    const float* bx = sel_boxes + ((size_t)(part * NMS_PRE + i)) * 7;
    float cx = bx[0], cz = bx[2], w = bx[3], h = bx[5], ry = bx[6];
    s_cx[i] = cx; s_cz[i] = cz; s_w[i] = w; s_h[i] = h;
    s_co[i] = cosf(ry); s_si[i] = sinf(ry);        // amortized: 512 per block total
    s_sq[i]  = cx * cx + cz * cz;
    s_rad[i] = 0.5f * sqrtf(w * w + h * h);
    s_val[i] = (i < pre) ? sel_valid[part * NMS_PRE + i] : 0u;
    #pragma unroll
    for (int q = 0; q < 16; ++q) s_adj[i][q] = 0u;
  }
  __syncthreads();

  const int nt = (pre + 15) >> 4;
  const int ntiles = nt * nt;
  const int wave = tid >> 5;
  const int lane = tid & 31;
  const int nwaves = blockDim.x >> 5;

  for (int t = wave; t < ntiles; t += nwaves) {   // uniform per wave
    int ti = t / nt, tj = t % nt;
    // Build A (rows, K=0..1 = cx,cz) and B (cols) per the 16-bit WMMA layouts:
    // lanes 0-15 hold K=0,1 in halves 0,1; lanes 16-31 (K=8..15/24..31) stay zero.
    v16h a = {}; v16h bm = {};
    if (lane < 16) {
      int ib = ti * 16 + lane;
      int jb = tj * 16 + lane;
      a[0]  = (_Float16)s_cx[ib]; a[1]  = (_Float16)s_cz[ib];
      bm[0] = (_Float16)s_cx[jb]; bm[1] = (_Float16)s_cz[jb];
    }
    v8f d = {};
    d = __builtin_amdgcn_wmma_f32_16x16x32_f16(false, a, false, bm,
                                               (short)0, d, false, false);
    // D layout: vgpr RR, lane -> M = RR + 8*(lane>=16), N = lane&15
    int jcol = tj * 16 + (lane & 15);
    int roff = (lane >= 16) ? 8 : 0;

    // literal-index accumulator reads (no movrel / M0 indexing)
    #define DO_PAIR(RR)                                                        \
    {                                                                          \
      int irow = ti * 16 + (RR) + roff;                                        \
      if (irow < pre && jcol < pre) {                                          \
        float dot = d[RR];                                                     \
        float dist2 = s_sq[irow] + s_sq[jcol] - 2.0f * dot;                    \
        float rad = s_rad[irow] + s_rad[jcol] + 2.0f; /* f16 error margin */   \
        if (dist2 <= rad * rad) {                                              \
          float iou = pair_iou_cs(s_cx[irow], s_cz[irow], s_w[irow],           \
                                  s_h[irow], s_co[irow], s_si[irow],           \
                                  s_cx[jcol], s_cz[jcol], s_w[jcol],           \
                                  s_h[jcol], s_co[jcol], s_si[jcol]);          \
          if (iou > NMS_THR)                                                   \
            atomicOr(&s_adj[irow][jcol >> 5], 1u << (jcol & 31));              \
        }                                                                      \
      }                                                                        \
    }
    DO_PAIR(0) DO_PAIR(1) DO_PAIR(2) DO_PAIR(3)
    DO_PAIR(4) DO_PAIR(5) DO_PAIR(6) DO_PAIR(7)
    #undef DO_PAIR
  }
  __syncthreads();

  if (tid == 0) {   // inherently sequential greedy suppression
    unsigned supp[16];
    #pragma unroll
    for (int q = 0; q < 16; ++q) supp[q] = 0u;
    int cnt = 0;
    for (int i = 0; i < pre; ++i) {
      bool sup = (supp[i >> 5] >> (i & 31)) & 1u;
      if (s_val[i] && !sup) {
        #pragma unroll
        for (int q = 0; q < 16; ++q) supp[q] |= s_adj[i][q];
        if (cnt < post) {
          const float* bx = sel_boxes + ((size_t)(part * NMS_PRE + i)) * 7;
          float* dst = stage_boxes + ((size_t)(part * NMS_POST + cnt)) * 7;
          #pragma unroll
          for (int c = 0; c < 7; ++c) dst[c] = bx[c];
          stage_scores[part * NMS_POST + cnt] = sel_scores[part * NMS_PRE + i];
        }
        cnt++;
      }
    }
    counts[part] = cnt < post ? cnt : post;
  }
}

/* ---------------- Kernel 4: concatenate range partitions -> d_out ------------- */
__global__ void gather_kernel(const float* __restrict__ stage_boxes,
                              const float* __restrict__ stage_scores,
                              const int* __restrict__ counts,
                              float* __restrict__ out) {
  const int b = blockIdx.x;
  const int k = threadIdx.x;   // 0..127
  int c0 = counts[b * 2 + 0];
  int c1 = counts[b * 2 + 1];
  float box[7] = {0, 0, 0, 0, 0, 0, 0};
  float sc = 0.0f;
  int src = -1;
  if (k < c0)            src = (b * 2 + 0) * NMS_POST + k;
  else if (k < c0 + c1)  src = (b * 2 + 1) * NMS_POST + (k - c0);
  if (src >= 0) {
    #pragma unroll
    for (int c = 0; c < 7; ++c) box[c] = stage_boxes[(size_t)src * 7 + c];
    sc = stage_scores[src];
  }
  float* ob = out + ((size_t)(b * NMS_POST + k)) * 7;
  #pragma unroll
  for (int c = 0; c < 7; ++c) ob[c] = box[c];
  out[(size_t)BB * NMS_POST * 7 + b * NMS_POST + k] = sc;
}

/* ---------------- host launch -------------------------------------------------- */
static inline size_t alignup(size_t x) { return (x + 255) & ~(size_t)255; }

extern "C" void kernel_launch(void* const* d_in, const int* in_sizes, int n_in,
                              void* d_out, int out_size, void* d_ws, size_t ws_size,
                              hipStream_t stream) {
  const float* rpn_scores = (const float*)d_in[0];   // (16, 16384)
  const float* rpn_reg    = (const float*)d_in[1];   // (16, 16384, 76)
  const float* xyz        = (const float*)d_in[2];   // (16, 16384, 3)
  const float* mean_size  = (const float*)d_in[3];   // (3,)
  float* out = (float*)d_out;                        // 16*128*7 + 16*128 floats

  char* ws = (char*)d_ws;
  float* boxes      = (float*)ws; ws += alignup((size_t)BB * NN * 7 * 4);
  float* sel_boxes  = (float*)ws; ws += alignup((size_t)BB * 2 * NMS_PRE * 7 * 4);
  float* sel_scores = (float*)ws; ws += alignup((size_t)BB * 2 * NMS_PRE * 4);
  unsigned* sel_valid = (unsigned*)ws; ws += alignup((size_t)BB * 2 * NMS_PRE * 4);
  float* stage_boxes  = (float*)ws; ws += alignup((size_t)BB * 2 * NMS_POST * 7 * 4);
  float* stage_scores = (float*)ws; ws += alignup((size_t)BB * 2 * NMS_POST * 4);
  int* counts = (int*)ws; ws += alignup((size_t)BB * 2 * 4);

  decode_kernel<<<dim3((BB * NN + 255) / 256), dim3(256), 0, stream>>>(
      rpn_reg, xyz, mean_size, boxes);

  // 128 KB dynamic LDS per workgroup (CDNA5: up to 320 KB/WGP)
  topk_kernel<<<dim3(BB * 2), dim3(1024), (size_t)NN * 8, stream>>>(
      rpn_scores, boxes, sel_boxes, sel_scores, sel_valid);

  nms_kernel<<<dim3(BB * 2), dim3(256), 0, stream>>>(
      sel_boxes, sel_scores, sel_valid, stage_boxes, stage_scores, counts);

  gather_kernel<<<dim3(BB), dim3(NMS_POST), 0, stream>>>(
      stage_boxes, stage_scores, counts, out);
}
